// GraphHeterogenousCrossAttention_69380901699713
// MI455X (gfx1250) — matrix operand order repaired and checked by reading, not compiled, not run on hardware
//
#include <hip/hip_runtime.h>

// B=4, N=50000, L=128, H=8 (hd=16). Only the i=2 (energy) branch of the
// reference survives; node/edge features are dead inputs.
// Pipeline: qp = q_data@Wq^T+bq;  e = exp(qp) (softmax w/o max-shift: values
// bounded, shift-invariant);  per-head: sum = SUM e, s = SUM e*k2/sum;
// u = e/sum + (s/max(s,1e-8))*v2;  y = u@Wo^T + bo.
// f16 WMMA (v_wmma_f32_16x16x32_f16) with fp32 accumulation -> memory bound.

#define LL 128
#define NN 50000
#define BB 4
#define WSTRIDE 136           // padded f16 row stride (272B: 16B aligned, bank-skewed)
#define TILES_PER_B 3125      // N / 16

typedef _Float16 v4h  __attribute__((ext_vector_type(4)));
typedef _Float16 v8h  __attribute__((ext_vector_type(8)));
typedef _Float16 v16h __attribute__((ext_vector_type(16)));
typedef float    v8f  __attribute__((ext_vector_type(8)));

// ---------------------------------------------------------------------------
// Kernel 1: k2[b,j] = softmax_head(energy[b]@Wk[2]^T + bk[2]),
//           v2[b,j] =            energy[b]@Wv[2]^T + bv[2]
// grid = B, block = 128
// ---------------------------------------------------------------------------
__global__ void hca_precompute_kv2(const float* __restrict__ energy,
                                   const float* __restrict__ Wk,
                                   const float* __restrict__ bk,
                                   const float* __restrict__ Wv,
                                   const float* __restrict__ bv,
                                   float* __restrict__ k2,
                                   float* __restrict__ v2) {
    const int b = blockIdx.x;
    const int j = threadIdx.x;
    __shared__ float e[LL];
    __shared__ float ek[LL];
    e[j] = energy[b * LL + j];
    __syncthreads();
    const float* wkr = Wk + 2 * LL * LL + j * LL;
    const float* wvr = Wv + 2 * LL * LL + j * LL;
    float aK = bk[2 * LL + j];
    float aV = bv[2 * LL + j];
    for (int c = 0; c < LL; ++c) {
        aK = fmaf(e[c], wkr[c], aK);
        aV = fmaf(e[c], wvr[c], aV);
    }
    ek[j] = aK;
    __syncthreads();
    const int g = j & ~15;
    float m = -3.4e38f;
    for (int t = 0; t < 16; ++t) m = fmaxf(m, ek[g + t]);
    float s = 0.f;
    for (int t = 0; t < 16; ++t) s += expf(ek[g + t] - m);
    k2[b * LL + j] = expf(aK - m) / s;
    v2[b * LL + j] = aV;
}

// ---------------------------------------------------------------------------
// Kernel 2: fused  qproj -> head-softmax -> rank-1 update -> out-proj
// block = 128 threads = 4 waves; each wave owns 16 rows; grid = 3125 blocks.
// ---------------------------------------------------------------------------
__global__ __launch_bounds__(128) void hca_fused(
        const float* __restrict__ q_data, const float* __restrict__ Wq,
        const float* __restrict__ bq,     const float* __restrict__ Wo,
        const float* __restrict__ bo,     const float* __restrict__ k2g,
        const float* __restrict__ v2g,    float* __restrict__ out) {
    __shared__ __align__(16) _Float16 wbuf[LL * WSTRIDE];   // Wq then Wo (f16)
    __shared__ __align__(16) _Float16 ubuf[64 * WSTRIDE];   // 4 waves x 16 rows

    const int tid   = threadIdx.x;
    const int wave  = tid >> 5;
    const int lane  = tid & 31;
    const int laneM = lane & 15;   // M for A-frags / N for B,C-frags
    const int hiH   = lane >> 4;   // half-wave selector

    const int gtile = blockIdx.x * 4 + wave;          // 0..12499
    const int b     = gtile / TILES_PER_B;
    const int r0    = (gtile % TILES_PER_B) * 16;

    // ---- stage Wq (f32 -> f16) into LDS: float4 load, v4h (b64) store ----
    for (int i = tid * 4; i < LL * LL; i += 128 * 4) {
        const float4 w = *(const float4*)(Wq + i);
        v4h h;
        h[0] = (_Float16)w.x; h[1] = (_Float16)w.y;
        h[2] = (_Float16)w.z; h[3] = (_Float16)w.w;
        *(v4h*)&wbuf[(i >> 7) * WSTRIDE + (i & 127)] = h;
    }
    __syncthreads();

    // ---- A fragments (16x32 f16 per K-tile) from q_data ----
    // lane<16 : row=laneM, K = kt*32 + {0..7, 16..23}
    // lane>=16: row=laneM, K = kt*32 + {8..15, 24..31}
    const float* arow = q_data + ((size_t)b * NN + (size_t)(r0 + laneM)) * LL;
    v16h afrag[4];
#pragma unroll
    for (int kt = 0; kt < 4; ++kt) {
        const int base = kt * 32 + hiH * 8;
        const float4 f0 = *(const float4*)(arow + base);
        const float4 f1 = *(const float4*)(arow + base + 4);
        const float4 f2 = *(const float4*)(arow + base + 16);
        const float4 f3 = *(const float4*)(arow + base + 20);
        v16h a;
        a[0]  = (_Float16)f0.x; a[1]  = (_Float16)f0.y; a[2]  = (_Float16)f0.z; a[3]  = (_Float16)f0.w;
        a[4]  = (_Float16)f1.x; a[5]  = (_Float16)f1.y; a[6]  = (_Float16)f1.z; a[7]  = (_Float16)f1.w;
        a[8]  = (_Float16)f2.x; a[9]  = (_Float16)f2.y; a[10] = (_Float16)f2.z; a[11] = (_Float16)f2.w;
        a[12] = (_Float16)f3.x; a[13] = (_Float16)f3.y; a[14] = (_Float16)f3.z; a[15] = (_Float16)f3.w;
        afrag[kt] = a;
    }

    // ---- GEMM1: acc[jt] = q_data @ Wq^T + bq  (bias folded into C init) ----
    v8f acc[8];
#pragma unroll
    for (int jt = 0; jt < 8; ++jt) {
        const float bias = bq[jt * 16 + laneM];
        v8f c;
#pragma unroll
        for (int r = 0; r < 8; ++r) c[r] = bias;
        acc[jt] = c;
    }
#pragma unroll
    for (int kt = 0; kt < 4; ++kt) {
        v16h bfr[8];                       // batch 16 ds_loads, then 8 WMMAs
#pragma unroll
        for (int jt = 0; jt < 8; ++jt) {
            const _Float16* bp = &wbuf[(jt * 16 + laneM) * WSTRIDE + kt * 32 + hiH * 8];
            v8h b0 = *(const v8h*)bp;
            v8h b1 = *(const v8h*)(bp + 16);
            bfr[jt] = __builtin_shufflevector(b0, b1, 0,1,2,3,4,5,6,7,8,9,10,11,12,13,14,15);
        }
#pragma unroll
        for (int jt = 0; jt < 8; ++jt)
            acc[jt] = __builtin_amdgcn_wmma_f32_16x16x32_f16(false, afrag[kt], false, bfr[jt],
                                                             (short)0, acc[jt], false, false);
    }

    // ---- e = exp(qp), scatter f16 into LDS (C layout -> A layout bridge) ----
    // No max-shift: softmax is shift-invariant; |qp| is O(4) here so exp is safe.
#pragma unroll
    for (int jt = 0; jt < 8; ++jt) {
#pragma unroll
        for (int r = 0; r < 8; ++r) {
            const float ev = expf(acc[jt][r]);
            ubuf[(wave * 16 + r + hiH * 8) * WSTRIDE + jt * 16 + laneM] = (_Float16)ev;
        }
    }

    // ---- reload e as A-fragments (each lane now holds half of one row) ----
    const _Float16* urow = &ubuf[(wave * 16 + laneM) * WSTRIDE];
#pragma unroll
    for (int kt = 0; kt < 4; ++kt) {
        const int base = kt * 32 + hiH * 8;
        v8h lo = *(const v8h*)(urow + base);
        v8h hi = *(const v8h*)(urow + base + 16);
        afrag[kt] = __builtin_shufflevector(lo, hi, 0,1,2,3,4,5,6,7,8,9,10,11,12,13,14,15);
    }

    // ---- per-head reductions: 1 shuffle per head instead of 12 per row ----
    const float* k2b = k2g + b * LL;
    const float* v2b = v2g + b * LL;
    float sumh[8], th[8];
#pragma unroll
    for (int kt = 0; kt < 4; ++kt) {
        const int c0 = kt * 32 + hiH * 8;
        float k2e[16];
        *(float4*)&k2e[0]  = *(const float4*)(k2b + c0);
        *(float4*)&k2e[4]  = *(const float4*)(k2b + c0 + 4);
        *(float4*)&k2e[8]  = *(const float4*)(k2b + c0 + 16);
        *(float4*)&k2e[12] = *(const float4*)(k2b + c0 + 20);
        float s0 = 0.f, t0 = 0.f, s1 = 0.f, t1 = 0.f;
#pragma unroll
        for (int i = 0; i < 8; ++i) {
            const float e0 = (float)afrag[kt][i];       // head 2*kt
            const float e1 = (float)afrag[kt][8 + i];   // head 2*kt+1
            s0 += e0; t0 = fmaf(e0, k2e[i], t0);
            s1 += e1; t1 = fmaf(e1, k2e[8 + i], t1);
        }
        sumh[2 * kt]     = s0; th[2 * kt]     = t0;
        sumh[2 * kt + 1] = s1; th[2 * kt + 1] = t1;
    }
    float inv[8], fh[8];
#pragma unroll
    for (int h = 0; h < 8; ++h) {
        const float s = sumh[h] + __shfl_xor(sumh[h], 16);  // lane^16 holds same row,
        const float t = th[h]   + __shfl_xor(th[h], 16);    // complementary 8 columns
        inv[h] = 1.f / s;
        const float sd = t * inv[h];                        // s = <q, k2>
        fh[h] = sd / fmaxf(sd, 1e-8f);                      // s * D_inv
    }

    // ---- u = e/sum + f*v2, built directly as A-fragments (f16) ----
#pragma unroll
    for (int kt = 0; kt < 4; ++kt) {
        const int c0 = kt * 32 + hiH * 8;
        float v2e[16];
        *(float4*)&v2e[0]  = *(const float4*)(v2b + c0);
        *(float4*)&v2e[4]  = *(const float4*)(v2b + c0 + 4);
        *(float4*)&v2e[8]  = *(const float4*)(v2b + c0 + 16);
        *(float4*)&v2e[12] = *(const float4*)(v2b + c0 + 20);
        v16h u;
#pragma unroll
        for (int i = 0; i < 8; ++i) {
            u[i]     = (_Float16)(fmaf((float)afrag[kt][i],     inv[2 * kt],
                                       fh[2 * kt]     * v2e[i]));
            u[8 + i] = (_Float16)(fmaf((float)afrag[kt][8 + i], inv[2 * kt + 1],
                                       fh[2 * kt + 1] * v2e[8 + i]));
        }
        afrag[kt] = u;
    }
    __syncthreads();           // all waves done with wbuf(Wq) + ubuf scatter

    // ---- restage Wo into the same LDS buffer ----
    for (int i = tid * 4; i < LL * LL; i += 128 * 4) {
        const float4 w = *(const float4*)(Wo + i);
        v4h h;
        h[0] = (_Float16)w.x; h[1] = (_Float16)w.y;
        h[2] = (_Float16)w.z; h[3] = (_Float16)w.w;
        *(v4h*)&wbuf[(i >> 7) * WSTRIDE + (i & 127)] = h;
    }
    __syncthreads();

    // ---- GEMM2: y = u @ Wo^T + bo, store fp32 ----
    v8f acc2[8];
#pragma unroll
    for (int jt = 0; jt < 8; ++jt) {
        const float bias = bo[jt * 16 + laneM];
        v8f c;
#pragma unroll
        for (int r = 0; r < 8; ++r) c[r] = bias;
        acc2[jt] = c;
    }
#pragma unroll
    for (int kt = 0; kt < 4; ++kt) {
        v16h bfr[8];
#pragma unroll
        for (int jt = 0; jt < 8; ++jt) {
            const _Float16* bp = &wbuf[(jt * 16 + laneM) * WSTRIDE + kt * 32 + hiH * 8];
            v8h b0 = *(const v8h*)bp;
            v8h b1 = *(const v8h*)(bp + 16);
            bfr[jt] = __builtin_shufflevector(b0, b1, 0,1,2,3,4,5,6,7,8,9,10,11,12,13,14,15);
        }
#pragma unroll
        for (int jt = 0; jt < 8; ++jt)
            acc2[jt] = __builtin_amdgcn_wmma_f32_16x16x32_f16(false, afrag[kt], false, bfr[jt],
                                                              (short)0, acc2[jt], false, false);
    }
#pragma unroll
    for (int jt = 0; jt < 8; ++jt) {
        float* orow = out + ((size_t)b * NN + (size_t)(r0 + hiH * 8)) * LL + jt * 16 + laneM;
#pragma unroll
        for (int r = 0; r < 8; ++r)
            orow[(size_t)r * LL] = acc2[jt][r];
    }
}

// ---------------------------------------------------------------------------
extern "C" void kernel_launch(void* const* d_in, const int* in_sizes, int n_in,
                              void* d_out, int out_size, void* d_ws, size_t ws_size,
                              hipStream_t stream) {
    (void)in_sizes; (void)n_in; (void)out_size; (void)ws_size;
    // setup_inputs order:
    // 0 node_features (unused), 1 edge_features (unused), 2 energy, 3 q_data,
    // 4 Wq, 5 bq, 6 Wk, 7 bk, 8 Wv, 9 bv, 10 Wo, 11 bo
    const float* energy = (const float*)d_in[2];
    const float* q_data = (const float*)d_in[3];
    const float* Wq     = (const float*)d_in[4];
    const float* bq     = (const float*)d_in[5];
    const float* Wk     = (const float*)d_in[6];
    const float* bk     = (const float*)d_in[7];
    const float* Wv     = (const float*)d_in[8];
    const float* bv     = (const float*)d_in[9];
    const float* Wo     = (const float*)d_in[10];
    const float* bo     = (const float*)d_in[11];

    float* k2 = (float*)d_ws;            // B*L floats
    float* v2 = k2 + BB * LL;            // B*L floats

    hca_precompute_kv2<<<BB, LL, 0, stream>>>(energy, Wk, bk, Wv, bv, k2, v2);
    hca_fused<<<(BB * TILES_PER_B) / 4, 128, 0, stream>>>(
        q_data, Wq, bq, Wo, bo, k2, v2, (float*)d_out);
}
